// OrthoInitPhasor_37082747633889
// MI455X (gfx1250) — compile-verified, matrix operand order checked by reading, not run on hardware
//
#include <hip/hip_runtime.h>
#include <hip/hip_bf16.h>
#include <math.h>

typedef __attribute__((ext_vector_type(16))) _Float16 v16h;
typedef __attribute__((ext_vector_type(8)))  _Float16 v8h;
typedef __attribute__((ext_vector_type(8)))  float    v8f;
typedef __attribute__((ext_vector_type(4)))  int      v4i;
typedef __attribute__((ext_vector_type(8)))  int      v8i;
typedef __attribute__((ext_vector_type(4)))  unsigned int v4u;

#define D_DIM 512
#define S_DIM 4096
#define B_DIM 4
#define CHUNK_LEN 64
#define MROWS (B_DIM * S_DIM)   // 16384

// GEMM tiling: block computes 128x64 of C, 8 waves, each wave a 16x64 strip.
#define BM 128
#define BN 64
#define BK 32
#define KDIM 512
#define LDT 48            // padded LDS stride for A tiles (halves)
#define BROWB 1056        // B panel LDS row stride in bytes: 1024 + 32 pad (TDM pad)

#if __has_builtin(__builtin_amdgcn_global_load_async_to_lds_b128) && \
    __has_builtin(__builtin_amdgcn_s_wait_asynccnt)
#define HAVE_ASYNC_LDS 1
#else
#define HAVE_ASYNC_LDS 0
#endif

#if __has_builtin(__builtin_amdgcn_tensor_load_to_lds) && \
    __has_builtin(__builtin_amdgcn_s_wait_tensorcnt)
#define HAVE_TDM 1
#else
#define HAVE_TDM 0
#endif

// ---------------------------------------------------------------------------
// fp32 -> f16 conversion (grid-stride)
// ---------------------------------------------------------------------------
__global__ void cvt_f32_f16_kernel(const float* __restrict__ in,
                                   _Float16* __restrict__ out, int n) {
  int i = blockIdx.x * blockDim.x + threadIdx.x;
  int stride = gridDim.x * blockDim.x;
  for (; i < n; i += stride) out[i] = (_Float16)in[i];
}

// ---------------------------------------------------------------------------
// async global -> LDS staging (gfx1250 ASYNCcnt path)
// ---------------------------------------------------------------------------
#if HAVE_ASYNC_LDS
__device__ __forceinline__ void async_copy_b128(const _Float16* g, _Float16* l) {
  __builtin_amdgcn_global_load_async_to_lds_b128(
      (__attribute__((address_space(1))) v4i*)(void*)(g),
      (__attribute__((address_space(3))) v4i*)(void*)(l),
      0, 0);
}
#endif

__device__ __forceinline__ unsigned lds_offset_u32(const void* p) {
  return (unsigned)(unsigned long long)
      (__attribute__((address_space(3))) char*)(void*)(p);
}

// ---------------------------------------------------------------------------
// C[M,N] = A[M,K] @ W[N,K]^T + bias[N]   (torch Linear convention)
// mode 0: Cf = f32   mode 1: Ch = f16   mode 2: Cf = f32 + resid
// B weight panel (BN x K) DMA'd once per block by the Tensor Data Mover with
// per-row LDS padding; A tiles double-buffered via async-to-LDS.
// ---------------------------------------------------------------------------
__global__ __launch_bounds__(256)
void gemm_wmma_kernel(const _Float16* __restrict__ A,
                      const _Float16* __restrict__ W,
                      const float* __restrict__ bias,
                      float* Cf, _Float16* Ch, const float* resid,
                      int M, int N, int K, int mode)
{
  __shared__ _Float16 sA[2][BM * LDT];                        // 2 x 12.3 KB
  __shared__ __align__(32) unsigned char sBp[BN * BROWB];     // 67.6 KB weight panel

  const int tid  = threadIdx.x;
  const int wave = tid >> 5;
  const int lane = tid & 31;
  const int m0 = blockIdx.x * BM;
  const int n0 = blockIdx.y * BN;

  // ---- stage the whole BN x K weight panel once -----------------------------
#if HAVE_TDM
  if (wave == 0) {
    const unsigned long long gaddr =
        (unsigned long long)(const void*)(W + (size_t)n0 * K);
    const unsigned ldsa = lds_offset_u32(sBp);
    v4u g0;
    g0.x = 1u;                                        // count=1, user mode
    g0.y = ldsa;                                      // lds_addr (bytes)
    g0.z = (unsigned)(gaddr & 0xFFFFFFFFu);           // global_addr[31:0]
    g0.w = (unsigned)((gaddr >> 32) & 0x01FFFFFFu)    // global_addr[56:32]
           | (2u << 30);                              // type = 2 (image)
    v8i g1;
    g1[0] = (int)((1u << 16)      // data_size = 2 bytes
                | (1u << 20)      // pad_enable
                | (7u << 22)      // pad_interval: 256 DWORDs (one 1024B row)
                | (7u << 25));    // pad_amount: 8 DWORDs (32B) -> row stride 1056B
    g1[1] = (int)(512u << 16);    // tensor_dim0 = 512 (bits 79:48)
    g1[2] = (int)(512u << 16);    // tensor_dim1 = 512 (bits 111:80)
    g1[3] = (int)(512u << 16);    // tile_dim0   = 512 (bits 127:112)
    g1[4] = 64;                   // tile_dim1   = 64  (bits 143:128)
    g1[5] = 512;                  // tensor_dim0_stride = 512 (bits 207:160)
    g1[6] = 0;
    g1[7] = 0;
    v4i g2 = {0, 0, 0, 0};
    v4i g3 = {0, 0, 0, 0};
    v8i g4 = {0, 0, 0, 0, 0, 0, 0, 0};
    __builtin_amdgcn_tensor_load_to_lds(g0, g1, g2, g3, g4, 0);
    __builtin_amdgcn_s_wait_tensorcnt(0);
  }
#else
  // fallback: cooperative copy, 8 halves per chunk, with the same row padding
  for (int h = tid; h < BN * (KDIM / 8); h += 256) {
    const int row = h >> 6;           // KDIM/8 = 64 chunks per row
    const int c8  = h & 63;
    *(v8h*)(sBp + (size_t)row * BROWB + c8 * 16) =
        *(const v8h*)(W + (size_t)(n0 + row) * K + c8 * 8);
  }
#endif

  // ---- A-tile staging assignments: 2 x b128 per thread per K-step ----------
  const int arow = tid >> 1;          // 0..127
  const int akp  = (tid & 1) * 16;    // 0 or 16 halves (32B chunk)
  const _Float16* gA = A + (size_t)(m0 + arow) * K + akp;

  v8f acc[4] = {};
  const int nk = K / BK;
  const int fr = lane & 15;
  const int ks = (lane >> 4) * 16;

#if HAVE_ASYNC_LDS
  async_copy_b128(gA,     &sA[0][arow * LDT + akp]);
  async_copy_b128(gA + 8, &sA[0][arow * LDT + akp + 8]);

  for (int i = 0; i < nk - 1; ++i) {
    const int cur = i & 1;
    const int nxt = cur ^ 1;
    const int kk = (i + 1) * BK;
    async_copy_b128(gA + kk,     &sA[nxt][arow * LDT + akp]);
    async_copy_b128(gA + kk + 8, &sA[nxt][arow * LDT + akp + 8]);
    __builtin_amdgcn_s_wait_asynccnt(2);   // current tile done, next in flight
    __syncthreads();

    v16h a = *(const v16h*)&sA[cur][(wave * 16 + fr) * LDT + ks];
#pragma unroll
    for (int j = 0; j < 4; ++j) {
      v16h b = *(const v16h*)(sBp + (size_t)(j * 16 + fr) * BROWB
                                  + (size_t)(i * BK + ks) * 2);
      acc[j] = __builtin_amdgcn_wmma_f32_16x16x32_f16(
          false, a, false, b, (short)0, acc[j], false, false);
    }
    __syncthreads();
  }
  {
    const int i = nk - 1;
    __builtin_amdgcn_s_wait_asynccnt(0);
    __syncthreads();
    v16h a = *(const v16h*)&sA[i & 1][(wave * 16 + fr) * LDT + ks];
#pragma unroll
    for (int j = 0; j < 4; ++j) {
      v16h b = *(const v16h*)(sBp + (size_t)(j * 16 + fr) * BROWB
                                  + (size_t)(i * BK + ks) * 2);
      acc[j] = __builtin_amdgcn_wmma_f32_16x16x32_f16(
          false, a, false, b, (short)0, acc[j], false, false);
    }
  }
#else
  for (int i = 0; i < nk; ++i) {
    const int kk = i * BK;
    *(v16h*)&sA[0][arow * LDT + akp] = *(const v16h*)(gA + kk);
    __syncthreads();
    v16h a = *(const v16h*)&sA[0][(wave * 16 + fr) * LDT + ks];
#pragma unroll
    for (int j = 0; j < 4; ++j) {
      v16h b = *(const v16h*)(sBp + (size_t)(j * 16 + fr) * BROWB
                                  + (size_t)(kk + ks) * 2);
      acc[j] = __builtin_amdgcn_wmma_f32_16x16x32_f16(
          false, a, false, b, (short)0, acc[j], false, false);
    }
    __syncthreads();
  }
#endif

  // C/D layout: lane%16 = N column, VGPR r -> M = r + 8*(lane>=16)
  const int colL  = lane & 15;
  const int rbase = (lane >> 4) * 8;
#pragma unroll
  for (int j = 0; j < 4; ++j) {
    const int col = n0 + j * 16 + colL;
    const float bv = bias[col];
#pragma unroll
    for (int r = 0; r < 8; ++r) {
      const int m = m0 + wave * 16 + rbase + r;
      const size_t idx = (size_t)m * N + col;
      const float v = acc[j][r] + bv;
      if (mode == 0)      Cf[idx] = v;
      else if (mode == 1) Ch[idx] = (_Float16)v;
      else                Cf[idx] = v + resid[idx];
    }
  }
}

// ---------------------------------------------------------------------------
// Phasor binding + chunked cumulative memory + retrieval.
// ---------------------------------------------------------------------------
__global__ __launch_bounds__(256)
void chunk_kernel(const float* __restrict__ value,
                  const float* __restrict__ kplin,
                  const float* qplin,
                  const float* __restrict__ bp,
                  const float* __restrict__ mod_scale,
                  float* retrieved)
{
  const int nC = S_DIM / CHUNK_LEN;
  const int b = blockIdx.x / nC;
  const int c = blockIdx.x % nC;
  const float ms = mod_scale[0];
  const float inv_sqrt_d = rsqrtf((float)D_DIM);

  const size_t base  = ((size_t)b * S_DIM + (size_t)c * CHUNK_LEN) * D_DIM;
  const size_t sbase = (size_t)c * CHUNK_LEN * D_DIM;

  for (int d = threadIdx.x; d < D_DIM; d += blockDim.x) {
    float mr = 0.f, mi = 0.f;
#pragma unroll 4
    for (int s = 0; s < CHUNK_LEN; ++s) {
      const size_t idx = base + (size_t)s * D_DIM + d;
      const float bpv = bp[sbase + (size_t)s * D_DIM + d];
      const float kp = bpv + kplin[idx] * ms;
      const float qp = bpv + qplin[idx] * ms;
      const float v = value[idx];
      float sk, ck, sq, cq;
      __sincosf(kp, &sk, &ck);
      __sincosf(qp, &sq, &cq);
      mr += v * ck;
      mi += v * sk;
      retrieved[idx] = (mr * cq + mi * sq) * inv_sqrt_d;
    }
  }
}

// ---------------------------------------------------------------------------
// LayerNorm over D=512, one block (256 thr) per row; writes f16 for final GEMM
// ---------------------------------------------------------------------------
__global__ __launch_bounds__(256)
void ln_kernel(const float* __restrict__ R,
               const float* __restrict__ g,
               const float* __restrict__ be,
               _Float16* __restrict__ out)
{
  const size_t row = blockIdx.x;
  const float* x = R + row * D_DIM;
  const int t = threadIdx.x;
  const float v0 = x[t];
  const float v1 = x[t + 256];

  __shared__ float rs[256];
  __shared__ float rq[256];
  rs[t] = v0 + v1;
  rq[t] = v0 * v0 + v1 * v1;
  __syncthreads();
  for (int o = 128; o > 0; o >>= 1) {
    if (t < o) { rs[t] += rs[t + o]; rq[t] += rq[t + o]; }
    __syncthreads();
  }
  const float mu   = rs[0] * (1.0f / D_DIM);
  const float var  = rq[0] * (1.0f / D_DIM) - mu * mu;
  const float rstd = rsqrtf(var + 1e-5f);

  out[row * D_DIM + t]       = (_Float16)((v0 - mu) * rstd * g[t]       + be[t]);
  out[row * D_DIM + t + 256] = (_Float16)((v1 - mu) * rstd * g[t + 256] + be[t + 256]);
}

// ---------------------------------------------------------------------------
extern "C" void kernel_launch(void* const* d_in, const int* in_sizes, int n_in,
                              void* d_out, int out_size, void* d_ws, size_t ws_size,
                              hipStream_t stream) {
  (void)in_sizes; (void)n_in; (void)out_size; (void)ws_size;

  const float* x   = (const float*)d_in[0];
  const float* bp  = (const float*)d_in[1];
  const float* Wk  = (const float*)d_in[2];
  const float* bk  = (const float*)d_in[3];
  const float* Wv  = (const float*)d_in[4];
  const float* bv  = (const float*)d_in[5];
  const float* Wq  = (const float*)d_in[6];
  const float* bq  = (const float*)d_in[7];
  const float* Wkm = (const float*)d_in[8];
  const float* bkm = (const float*)d_in[9];
  const float* Wqm = (const float*)d_in[10];
  const float* bqm = (const float*)d_in[11];
  const float* ms  = (const float*)d_in[12];
  const float* lng = (const float*)d_in[13];
  const float* lnb = (const float*)d_in[14];
  const float* Wo  = (const float*)d_in[15];
  const float* bo  = (const float*)d_in[16];
  float* out = (float*)d_out;

  const size_t MSD = (size_t)MROWS * D_DIM;   // 8,388,608
  const int    DD  = D_DIM * D_DIM;           // 262,144

  char* p = (char*)d_ws;
  auto wsalloc = [&](size_t bytes) -> void* {
    void* r = (void*)p;
    p += (bytes + 255) & ~(size_t)255;
    return r;
  };
  _Float16* xh   = (_Float16*)wsalloc(MSD * 2);  // x (f16), later reused as `normed`
  _Float16* kh   = (_Float16*)wsalloc(MSD * 2);
  _Float16* qh   = (_Float16*)wsalloc(MSD * 2);
  float*    vf   = (float*)   wsalloc(MSD * 4);
  float*    kpf  = (float*)   wsalloc(MSD * 4);
  float*    qpf  = (float*)   wsalloc(MSD * 4);  // later aliased as `retrieved`
  _Float16* wkh  = (_Float16*)wsalloc((size_t)DD * 2);
  _Float16* wvh  = (_Float16*)wsalloc((size_t)DD * 2);
  _Float16* wqh  = (_Float16*)wsalloc((size_t)DD * 2);
  _Float16* wkmh = (_Float16*)wsalloc((size_t)DD * 2);
  _Float16* wqmh = (_Float16*)wsalloc((size_t)DD * 2);
  _Float16* woh  = (_Float16*)wsalloc((size_t)DD * 2);

  // --- precision conversion -------------------------------------------------
  cvt_f32_f16_kernel<<<2048, 256, 0, stream>>>(x, xh, (int)MSD);
  cvt_f32_f16_kernel<<<256, 256, 0, stream>>>(Wk,  wkh,  DD);
  cvt_f32_f16_kernel<<<256, 256, 0, stream>>>(Wv,  wvh,  DD);
  cvt_f32_f16_kernel<<<256, 256, 0, stream>>>(Wq,  wqh,  DD);
  cvt_f32_f16_kernel<<<256, 256, 0, stream>>>(Wkm, wkmh, DD);
  cvt_f32_f16_kernel<<<256, 256, 0, stream>>>(Wqm, wqmh, DD);
  cvt_f32_f16_kernel<<<256, 256, 0, stream>>>(Wo,  woh,  DD);

  const dim3 ggrid(MROWS / BM, D_DIM / BN);   // 128 x 8

  // --- stage 1: key / value / query ----------------------------------------
  gemm_wmma_kernel<<<ggrid, 256, 0, stream>>>(xh, wkh, bk, nullptr, kh, nullptr,
                                              MROWS, D_DIM, D_DIM, /*f16*/1);
  gemm_wmma_kernel<<<ggrid, 256, 0, stream>>>(xh, wvh, bv, vf, nullptr, nullptr,
                                              MROWS, D_DIM, D_DIM, /*f32*/0);
  gemm_wmma_kernel<<<ggrid, 256, 0, stream>>>(xh, wqh, bq, nullptr, qh, nullptr,
                                              MROWS, D_DIM, D_DIM, /*f16*/1);

  // --- stage 2: phase modulation linears -----------------------------------
  gemm_wmma_kernel<<<ggrid, 256, 0, stream>>>(kh, wkmh, bkm, kpf, nullptr, nullptr,
                                              MROWS, D_DIM, D_DIM, 0);
  gemm_wmma_kernel<<<ggrid, 256, 0, stream>>>(qh, wqmh, bqm, qpf, nullptr, nullptr,
                                              MROWS, D_DIM, D_DIM, 0);

  // --- stage 3: bind, chunked cumulative memory, retrieve (in-place in qpf)-
  chunk_kernel<<<B_DIM * (S_DIM / CHUNK_LEN), 256, 0, stream>>>(
      vf, kpf, qpf, bp, ms, /*retrieved=*/qpf);

  // --- stage 4: LayerNorm -> normed (f16, reuses xh) ------------------------
  ln_kernel<<<MROWS, 256, 0, stream>>>(qpf, lng, lnb, xh);

  // --- stage 5: to_out GEMM + residual -------------------------------------
  gemm_wmma_kernel<<<ggrid, 256, 0, stream>>>(xh, woh, bo, out, nullptr, x,
                                              MROWS, D_DIM, D_DIM, /*resid*/2);
}
